// TPFN_76836964925797
// MI455X (gfx1250) — compile-verified, matrix operand order checked by reading, not run on hardware
//
#include <hip/hip_runtime.h>
#include <math.h>

typedef __bf16 bf16_t;
typedef __attribute__((ext_vector_type(16))) __bf16 v16bf;
typedef __attribute__((ext_vector_type(8)))  __bf16 v8bf;
typedef __attribute__((ext_vector_type(8)))  float  v8f;
typedef __attribute__((ext_vector_type(4)))  unsigned int u32x4;
typedef __attribute__((ext_vector_type(8)))  int i32x8;
typedef __attribute__((ext_vector_type(4)))  int i32x4;

#ifndef __has_builtin
#define __has_builtin(x) 0
#endif
#if __has_builtin(__builtin_amdgcn_tensor_load_to_lds) && __has_builtin(__builtin_amdgcn_s_wait_tensorcnt)
#define USE_TDM 1
#else
#define USE_TDM 0
#endif

#define DEV __device__ __forceinline__

DEV v16bf cat16(v8bf lo, v8bf hi) {
  return __builtin_shufflevector(lo, hi, 0, 1, 2, 3, 4, 5, 6, 7,
                                 8, 9, 10, 11, 12, 13, 14, 15);
}

DEV v8f wmma_bf16f32(v16bf a, v16bf b, v8f c) {
  // D = A(16x32) * B(32x16) + C, f32 accumulate
  return __builtin_amdgcn_wmma_f32_16x16x32_bf16(
      false, a, false, b, (short)0, c, false, false);
}

// 8 consecutive fp32 -> bf16. D = true K extent; if D%32==0 all slabs are full
// and rows stay 16B aligned -> b128 loads, else per-element guard (pads zero).
template <int D>
DEV v8bf load8_f32(const float* __restrict__ row, int k) {
  v8bf r;
  if constexpr (D % 32 == 0) {
    const float4 a = *(const float4*)(row + k);
    const float4 b = *(const float4*)(row + k + 4);
    r[0] = (__bf16)a.x; r[1] = (__bf16)a.y; r[2] = (__bf16)a.z; r[3] = (__bf16)a.w;
    r[4] = (__bf16)b.x; r[5] = (__bf16)b.y; r[6] = (__bf16)b.z; r[7] = (__bf16)b.w;
  } else {
#pragma unroll
    for (int j = 0; j < 8; ++j) r[j] = (__bf16)((k + j < D) ? row[k + j] : 0.f);
  }
  return r;
}

template <int D>
DEV v16bf frag_f32(const float* __restrict__ row, int k0) {
  return cat16(load8_f32<D>(row, k0), load8_f32<D>(row, k0 + 16));
}

// bf16 source (global or LDS), 16B-aligned, unguarded.
DEV v16bf frag_bf16(const bf16_t* __restrict__ row, int k0) {
  v8bf lo = *(const v8bf*)(row + k0);
  v8bf hi = *(const v8bf*)(row + k0 + 16);
  return cat16(lo, hi);
}

DEV float sigm(float x) { return 1.f / (1.f + __expf(-x)); }

// ---- Tensor Data Mover: 2D strided tile (rows x tile_dw dwords) -> LDS ------
// D# per CDNA5 ISA 8.3/8.4: group0 = {count, lds_addr, global_addr, type=2},
// group1 = {data_size=4B, tensor_dim0/1, tile_dim0/1, tensor_dim0_stride}.
DEV void tdm_load_2d(unsigned lds_off, const void* gaddr, unsigned tile_dw,
                     unsigned rows, unsigned long long row_stride_dw) {
#if USE_TDM
  unsigned long long ga = (unsigned long long)(uintptr_t)gaddr;
  u32x4 g0;
  g0[0] = 1u;                                         // count=1, user mode
  g0[1] = lds_off;                                    // lds_addr (bytes)
  g0[2] = (unsigned)(ga & 0xFFFFFFFFu);               // global_addr[31:0]
  g0[3] = (unsigned)((ga >> 32) & 0x1FFFFFFu) | (2u << 30);  // addr[56:32], type=2
  i32x8 g1;
  g1[0] = (2 << 16);                                  // data_size = 4 bytes
  g1[1] = (int)((tile_dw & 0xFFFFu) << 16);           // tensor_dim0[15:0]
  g1[2] = (int)(((tile_dw >> 16) & 0xFFFFu) | ((rows & 0xFFFFu) << 16)); // dim0 hi, dim1 lo
  g1[3] = (int)((tile_dw & 0xFFFFu) << 16);           // dim1 hi=0, tile_dim0
  g1[4] = (int)(rows & 0xFFFFu);                      // tile_dim1, tile_dim2=0
  g1[5] = (int)(row_stride_dw & 0xFFFFFFFFull);       // tensor_dim0_stride[31:0]
  g1[6] = (int)((row_stride_dw >> 32) & 0xFFFFull);   // stride[47:32], dim1_stride=0
  g1[7] = 0;
  i32x4 gz = {0, 0, 0, 0};
#if defined(__clang_major__) && (__clang_major__ >= 23)
  i32x8 gz8 = {0, 0, 0, 0, 0, 0, 0, 0};
  __builtin_amdgcn_tensor_load_to_lds(g0, g1, gz, gz, gz8, 0);
#else
  __builtin_amdgcn_tensor_load_to_lds(g0, g1, gz, gz, 0);
#endif
#else
  (void)lds_off; (void)gaddr; (void)tile_dw; (void)rows; (void)row_stride_dw;
#endif
}

// s_wait_tensorcnt needs a compile-time constant SIMM16
template <int NCNT>
DEV void tdm_wait() {
#if USE_TDM
  __builtin_amdgcn_s_wait_tensorcnt(NCNT);
#endif
}

// ---------------------------------------------------------------- prep kernels
__global__ void cvt_f32_bf16(const float* __restrict__ in, bf16_t* __restrict__ out, int n) {
  int i = blockIdx.x * blockDim.x + threadIdx.x;
  if (i < n) out[i] = (bf16_t)in[i];
}

// dst [rows][Dp] bf16, src [rows][D] fp32, K zero-padded D -> Dp
__global__ void cvt_pad_rows(const float* __restrict__ src, bf16_t* __restrict__ dst,
                             int D, int Dp, int total) {
  int i = blockIdx.x * blockDim.x + threadIdx.x;
  if (i >= total) return;
  int r = i / Dp, k = i % Dp;
  dst[i] = (bf16_t)((k < D) ? src[(size_t)r * D + k] : 0.f);
}

// out [32][256] bf16, rows >= 24 zero-padded
__global__ void build_pad24(const float* __restrict__ W, bf16_t* __restrict__ out,
                            int srcStride, int srcOff) {
  int i = blockIdx.x * blockDim.x + threadIdx.x;
  if (i >= 32 * 256) return;
  int r = i >> 8, k = i & 255;
  out[i] = (bf16_t)((r < 24) ? W[(size_t)r * srcStride + srcOff + k] : 0.f);
}

// --------------------------------------------------- input-projection GEMM
// G[m,n] = sum_k X[m,k] * Wih[n,k] + bias[n]; 32 M-rows per block (2 M-tiles
// per wave, B-frag reused across both), 8 waves split N. Wb is bf16, K padded.
template <int D, int DP, int N>
__global__ __launch_bounds__(256) void gates_gemm(const float* __restrict__ X,
                                                  const bf16_t* __restrict__ Wb,
                                                  const float* __restrict__ bias,
                                                  bf16_t* __restrict__ G) {
  constexpr int NT = N / 128;   // n-tiles per wave
  constexpr int KS = DP / 32;   // K slabs
  constexpr int MT = 2;         // m-tiles per block
  const int row0  = blockIdx.x * (16 * MT);
  const int lane  = threadIdx.x & 31;
  const int wid   = threadIdx.x >> 5;
  const int nlane = lane & 15;
  const int kb    = (lane >> 4) * 8;
  const int mtop  = (lane >> 4) * 8;
  const int nbase = wid * NT * 16;

  const float* xrow[MT];
#pragma unroll
  for (int mt = 0; mt < MT; ++mt)
    xrow[mt] = X + (size_t)(row0 + mt * 16 + nlane) * D;

  v8f acc[MT][NT];
#pragma unroll
  for (int mt = 0; mt < MT; ++mt)
#pragma unroll
    for (int nt = 0; nt < NT; ++nt)
      acc[mt][nt] = (v8f){0.f, 0.f, 0.f, 0.f, 0.f, 0.f, 0.f, 0.f};

  for (int ks = 0; ks < KS; ++ks) {
    const int k0 = ks * 32 + kb;
    if (ks + 1 < KS) {
      __builtin_prefetch(xrow[0] + (ks + 1) * 32 + kb, 0, 1);
      __builtin_prefetch(xrow[1] + (ks + 1) * 32 + kb, 0, 1);
    }
    v16bf af[MT];
#pragma unroll
    for (int mt = 0; mt < MT; ++mt) af[mt] = frag_f32<D>(xrow[mt], k0);
#pragma unroll
    for (int nt = 0; nt < NT; ++nt) {
      const int n = nbase + nt * 16 + nlane;
      v16bf bf = frag_bf16(Wb + (size_t)n * DP, k0);
#pragma unroll
      for (int mt = 0; mt < MT; ++mt)
        acc[mt][nt] = wmma_bf16f32(af[mt], bf, acc[mt][nt]);
    }
  }
#pragma unroll
  for (int mt = 0; mt < MT; ++mt)
#pragma unroll
    for (int nt = 0; nt < NT; ++nt) {
      const int n  = nbase + nt * 16 + nlane;
      const float bv = bias[n];
#pragma unroll
      for (int r = 0; r < 8; ++r)
        G[(size_t)(row0 + mt * 16 + mtop + r) * N + n] = (bf16_t)(acc[mt][nt][r] + bv);
    }
}

// ------------------------------------------------------- sequential LSTM
// One block = 16 batch rows; loops T=256 steps. Recurrent GEMM via WMMA from
// LDS-resident h; per-step precomputed gate tile (16 rows, stride T*4H) is
// TDM-double-buffered into LDS one step ahead (s_wait_tensorcnt to sync).
template <int H>
DEV void lstm_body(const bf16_t* __restrict__ G, const bf16_t* __restrict__ Whh,
                   bf16_t* __restrict__ Z, int colOff,
                   bf16_t (*hS)[128], float (*cS)[128], bf16_t (*gStage)[512],
                   bf16_t* gbuf /* [2][16*4H] */) {
  constexpr int T = 256, LEN = 256;
  constexpr int N = 4 * H;
  constexpr int NT = N / 128;
  constexpr int KS = H / 32;
  const int b0    = blockIdx.x * 16;
  const int tid   = threadIdx.x;
  const int lane  = tid & 31;
  const int wid   = tid >> 5;
  const int nlane = lane & 15;
  const int kb    = (lane >> 4) * 8;
  const int mtop  = (lane >> 4) * 8;
  const int nbase = wid * NT * 16;

  for (int i = tid; i < 16 * 128; i += 256) {
    (&hS[0][0])[i] = (bf16_t)0.f;
    (&cS[0][0])[i] = 0.f;
  }
  __syncthreads();

#if USE_TDM
  // prologue: fetch gate tile for t=0 into buffer 0
  if (wid == 0) {
    tdm_load_2d((unsigned)(uintptr_t)gbuf, G + ((size_t)b0 * T + 0) * N,
                N / 2, 16, (unsigned long long)T * (N / 2));
  }
#endif

  for (int t = 0; t < T; ++t) {
#if USE_TDM
    if (wid == 0 && t + 1 < T) {   // prefetch next step while we do the GEMM
      tdm_load_2d((unsigned)(uintptr_t)(gbuf + ((t + 1) & 1) * 16 * N),
                  G + ((size_t)b0 * T + (t + 1)) * N,
                  N / 2, 16, (unsigned long long)T * (N / 2));
    }
#endif
    // ---- WMMA: gStage = hS @ Whh^T
#pragma unroll
    for (int nt = 0; nt < NT; ++nt) {
      const int n = nbase + nt * 16 + nlane;
      v8f acc = (v8f){0.f, 0.f, 0.f, 0.f, 0.f, 0.f, 0.f, 0.f};
#pragma unroll
      for (int ks = 0; ks < KS; ++ks) {
        const int k0 = ks * 32 + kb;
        v16bf af = frag_bf16(&hS[nlane][0], k0);
        v16bf bf = frag_bf16(Whh + (size_t)n * H, k0);
        acc = wmma_bf16f32(af, bf, acc);
      }
#pragma unroll
      for (int r = 0; r < 8; ++r) gStage[mtop + r][n] = (bf16_t)acc[r];
    }
#if USE_TDM
    if (wid == 0) {                 // tile t resident before the cell update
      if (t + 1 < T) tdm_wait<1>(); // keep the t+1 prefetch in flight
      else           tdm_wait<0>(); // last step: drain
    }
#endif
    __syncthreads();

    // ---- cell update (i,f,g,o), h -> LDS, z -> global
    constexpr int CELLS = 16 * H;
    for (int cidx = tid; cidx < CELLS; cidx += 256) {
      const int bl = cidx / H, j = cidx % H;
      const size_t row = (size_t)(b0 + bl) * T + t;
#if USE_TDM
      const bf16_t* gp = gbuf + (t & 1) * 16 * N + bl * N;
#else
      const bf16_t* gp = G + row * N;
#endif
      const float ig = (float)gStage[bl][j]         + (float)gp[j];
      const float fg = (float)gStage[bl][H + j]     + (float)gp[H + j];
      const float gg = (float)gStage[bl][2 * H + j] + (float)gp[2 * H + j];
      const float og = (float)gStage[bl][3 * H + j] + (float)gp[3 * H + j];
      const float c = sigm(fg) * cS[bl][j] + sigm(ig) * tanhf(gg);
      const float h = sigm(og) * tanhf(c);
      cS[bl][j] = c;
      hS[bl][j] = (bf16_t)h;
      Z[row * LEN + colOff + j] = (bf16_t)h;
    }
    __syncthreads();
  }
}

// all three modalities in one launch: grid=(16,3) -> 48 concurrent blocks
__global__ __launch_bounds__(256) void lstm_all(const bf16_t* __restrict__ gA,
                                                const bf16_t* __restrict__ gV,
                                                const bf16_t* __restrict__ gT,
                                                const bf16_t* __restrict__ whhA,
                                                const bf16_t* __restrict__ whhV,
                                                const bf16_t* __restrict__ whhT,
                                                bf16_t* __restrict__ Z) {
  __shared__ alignas(16) bf16_t hS[16][128];
  __shared__ float cS[16][128];
  __shared__ alignas(16) bf16_t gStage[16][512];
  __shared__ alignas(16) bf16_t gbuf[2 * 16 * 512];
  if (blockIdx.y == 0)      lstm_body<64>(gA, whhA, Z, 0,   hS, cS, gStage, gbuf);
  else if (blockIdx.y == 1) lstm_body<64>(gV, whhV, Z, 64,  hS, cS, gStage, gbuf);
  else                      lstm_body<128>(gT, whhT, Z, 128, hS, cS, gStage, gbuf);
}

// ------------------------------------------------------------ fusion GEMMs
// wave0: A  = z @ W1p^T + b1                  (rank 24 padded to N=32)
// wave1: Bb = z @ W2a^T + roll(z) @ W2b^T + (b2 + W2[:,0])
__global__ __launch_bounds__(64) void fusion_gemm(const bf16_t* __restrict__ Z,
                                                  const bf16_t* __restrict__ W1p,
                                                  const bf16_t* __restrict__ W2a,
                                                  const bf16_t* __restrict__ W2b,
                                                  const float* __restrict__ b1,
                                                  const float* __restrict__ b2,
                                                  const float* __restrict__ W2full,
                                                  float* __restrict__ Af,
                                                  float* __restrict__ Bf) {
  constexpr int T = 256, LEN = 256;
  const int row0  = blockIdx.x * 16;
  const int bidx  = row0 >> 8;
  const int t0    = row0 & 255;
  const int lane  = threadIdx.x & 31;
  const int wid   = threadIdx.x >> 5;
  const int nlane = lane & 15;
  const int kb    = (lane >> 4) * 8;
  const int mtop  = (lane >> 4) * 8;

  v8f acc0 = (v8f){0.f, 0.f, 0.f, 0.f, 0.f, 0.f, 0.f, 0.f};
  v8f acc1 = acc0;
  const int n0 = nlane, n1 = 16 + nlane;

  if (wid == 0) {
    const bf16_t* zrow = Z + (size_t)(row0 + nlane) * LEN;
    const bf16_t* w0 = W1p + (size_t)n0 * LEN;
    const bf16_t* w1 = W1p + (size_t)n1 * LEN;
#pragma unroll
    for (int ks = 0; ks < 8; ++ks) {
      const int k0 = ks * 32 + kb;
      v16bf af = frag_bf16(zrow, k0);
      acc0 = wmma_bf16f32(af, frag_bf16(w0, k0), acc0);
      acc1 = wmma_bf16f32(af, frag_bf16(w1, k0), acc1);
    }
    const float bv0 = (n0 < 24) ? b1[n0] : 0.f;
    const float bv1 = (n1 < 24) ? b1[n1] : 0.f;
#pragma unroll
    for (int r = 0; r < 8; ++r) {
      Af[(size_t)(row0 + mtop + r) * 32 + n0] = acc0[r] + bv0;
      Af[(size_t)(row0 + mtop + r) * 32 + n1] = acc1[r] + bv1;
    }
  } else {
    const int tA = t0 + nlane;        // A-row M = lane%16 -> t = t0+M
    const int tB = (tA + 1) & 255;    // time roll with wrap
    const bf16_t* zrow  = Z + ((size_t)bidx * T + tA) * LEN;
    const bf16_t* zrow2 = Z + ((size_t)bidx * T + tB) * LEN;
    const bf16_t* wa0 = W2a + (size_t)n0 * LEN;
    const bf16_t* wa1 = W2a + (size_t)n1 * LEN;
    const bf16_t* wb0 = W2b + (size_t)n0 * LEN;
    const bf16_t* wb1 = W2b + (size_t)n1 * LEN;
#pragma unroll
    for (int ks = 0; ks < 8; ++ks) {
      const int k0 = ks * 32 + kb;
      v16bf af  = frag_bf16(zrow, k0);
      acc0 = wmma_bf16f32(af, frag_bf16(wa0, k0), acc0);
      acc1 = wmma_bf16f32(af, frag_bf16(wa1, k0), acc1);
      v16bf af2 = frag_bf16(zrow2, k0);
      acc0 = wmma_bf16f32(af2, frag_bf16(wb0, k0), acc0);
      acc1 = wmma_bf16f32(af2, frag_bf16(wb1, k0), acc1);
    }
    const float bv0 = (n0 < 24) ? (b2[n0] + W2full[(size_t)n0 * 513]) : 0.f;
    const float bv1 = (n1 < 24) ? (b2[n1] + W2full[(size_t)n1 * 513]) : 0.f;
#pragma unroll
    for (int r = 0; r < 8; ++r) {
      Bf[(size_t)(row0 + mtop + r) * 32 + n0] = acc0[r] + bv0;
      Bf[(size_t)(row0 + mtop + r) * 32 + n1] = acc1[r] + bv1;
    }
  }
}

// ----------------------------------------------------------- reductions
__global__ __launch_bounds__(256) void reduce_bt(const float* __restrict__ Af,
                                                 const float* __restrict__ Bf,
                                                 const bf16_t* __restrict__ Z,
                                                 const float* __restrict__ fw,
                                                 float* __restrict__ outp,
                                                 float* __restrict__ normb) {
  const int b = blockIdx.x, tid = threadIdx.x;  // tid == t
  const size_t row = (size_t)b * 256 + tid;
  const float* ap = Af + row * 32;
  const float* bp = Bf + row * 32;
  float s = 0.f;
#pragma unroll
  for (int r = 0; r < 24; ++r) s += fw[r] * ap[r] * bp[r];
  float q = 0.f;
  const v8bf* zp = (const v8bf*)(Z + row * 256);
#pragma unroll 4
  for (int l = 0; l < 32; ++l) {
    v8bf v = zp[l];
#pragma unroll
    for (int j = 0; j < 8; ++j) { const float zv = (float)v[j]; q += zv * zv; }
  }
  __shared__ float sS[256], qS[256];
  sS[tid] = s; qS[tid] = q;
  __syncthreads();
  for (int st = 128; st > 0; st >>= 1) {
    if (tid < st) { sS[tid] += sS[tid + st]; qS[tid] += qS[tid + st]; }
    __syncthreads();
  }
  if (tid == 0) {
    outp[b]  = sS[0] * (1.f / 256.f);   // time mean folded with fw dot
    normb[b] = sqrtf(qS[0]);
  }
}

__global__ __launch_bounds__(256) void norm_final(const float* __restrict__ normb,
                                                  float* __restrict__ outp) {
  const int tid = threadIdx.x;
  __shared__ float sS[256];
  sS[tid] = normb[tid];
  __syncthreads();
  for (int st = 128; st > 0; st >>= 1) {
    if (tid < st) sS[tid] += sS[tid + st];
    __syncthreads();
  }
  if (tid == 0) outp[256] = sS[0] * (1.f / 256.f);
}

// ---------------------------------------------------------------- launcher
extern "C" void kernel_launch(void* const* d_in, const int* in_sizes, int n_in,
                              void* d_out, int out_size, void* d_ws, size_t ws_size,
                              hipStream_t stream) {
  (void)in_sizes; (void)n_in; (void)out_size; (void)ws_size;
  const float* text   = (const float*)d_in[0];
  const float* audio  = (const float*)d_in[1];
  const float* vision = (const float*)d_in[2];
  const float* Wih_t  = (const float*)d_in[3];
  const float* Whh_t  = (const float*)d_in[4];
  const float* b_t    = (const float*)d_in[5];
  const float* Wih_a  = (const float*)d_in[6];
  const float* Whh_a  = (const float*)d_in[7];
  const float* b_a    = (const float*)d_in[8];
  const float* Wih_v  = (const float*)d_in[9];
  const float* Whh_v  = (const float*)d_in[10];
  const float* b_v    = (const float*)d_in[11];
  const float* W1     = (const float*)d_in[12];
  const float* b1     = (const float*)d_in[13];
  const float* W2     = (const float*)d_in[14];
  const float* b2     = (const float*)d_in[15];
  const float* fw     = (const float*)d_in[16];
  float* out = (float*)d_out;

  const int BT = 256 * 256;
  char* ws = (char*)d_ws;
  size_t off = 0;
  auto take = [&](size_t bytes) -> char* {
    char* p = ws + off;
    off = (off + bytes + 255) & ~(size_t)255;
    return p;
  };
  bf16_t* wihT = (bf16_t*)take((size_t)512 * 768 * 2);
  bf16_t* wihA = (bf16_t*)take((size_t)256 * 96 * 2);
  bf16_t* wihV = (bf16_t*)take((size_t)256 * 64 * 2);
  bf16_t* whhT = (bf16_t*)take((size_t)512 * 128 * 2);
  bf16_t* whhA = (bf16_t*)take((size_t)256 * 64 * 2);
  bf16_t* whhV = (bf16_t*)take((size_t)256 * 64 * 2);
  bf16_t* w1p  = (bf16_t*)take((size_t)32 * 256 * 2);
  bf16_t* w2a  = (bf16_t*)take((size_t)32 * 256 * 2);
  bf16_t* w2b  = (bf16_t*)take((size_t)32 * 256 * 2);
  bf16_t* gT   = (bf16_t*)take((size_t)BT * 512 * 2);
  bf16_t* gA   = (bf16_t*)take((size_t)BT * 256 * 2);
  bf16_t* gV   = (bf16_t*)take((size_t)BT * 256 * 2);
  bf16_t* z    = (bf16_t*)take((size_t)BT * 256 * 2);
  float*  Afus = (float*)take((size_t)BT * 32 * 4);
  float*  Bfus = (float*)take((size_t)BT * 32 * 4);
  float*  normb = (float*)take((size_t)256 * 4);

  // prep: bf16 weight images (K-padded) + padded fusion weights
  cvt_pad_rows<<<(512 * 768 + 255) / 256, 256, 0, stream>>>(Wih_t, wihT, 768, 768, 512 * 768);
  cvt_pad_rows<<<(256 * 96 + 255) / 256, 256, 0, stream>>>(Wih_a, wihA, 74, 96, 256 * 96);
  cvt_pad_rows<<<(256 * 64 + 255) / 256, 256, 0, stream>>>(Wih_v, wihV, 35, 64, 256 * 64);
  cvt_f32_bf16<<<(512 * 128 + 255) / 256, 256, 0, stream>>>(Whh_t, whhT, 512 * 128);
  cvt_f32_bf16<<<(256 * 64 + 255) / 256, 256, 0, stream>>>(Whh_a, whhA, 256 * 64);
  cvt_f32_bf16<<<(256 * 64 + 255) / 256, 256, 0, stream>>>(Whh_v, whhV, 256 * 64);
  build_pad24<<<32, 256, 0, stream>>>(W1, w1p, 256, 0);
  build_pad24<<<32, 256, 0, stream>>>(W2, w2a, 513, 1);
  build_pad24<<<32, 256, 0, stream>>>(W2, w2b, 513, 257);

  // input projections (big WMMA GEMMs, 32 M-rows per block)
  gates_gemm<768, 768, 512><<<BT / 32, 256, 0, stream>>>(text, wihT, b_t, gT);
  gates_gemm<74, 96, 256><<<BT / 32, 256, 0, stream>>>(audio, wihA, b_a, gA);
  gates_gemm<35, 64, 256><<<BT / 32, 256, 0, stream>>>(vision, wihV, b_v, gV);

  // sequential LSTMs -> z (concat order: audio, video, text), fused launch
  lstm_all<<<dim3(16, 3), 256, 0, stream>>>(gA, gV, gT, whhA, whhV, whhT, z);

  // fusion projections + reductions
  fusion_gemm<<<BT / 16, 64, 0, stream>>>(z, w1p, w2a, w2b, b1, b2, W2, Afus, Bfus);
  reduce_bt<<<256, 256, 0, stream>>>(Afus, Bfus, z, fw, out, normb);
  norm_final<<<1, 256, 0, stream>>>(normb, out);
}
